// RNNDecoder_85160611545333
// MI455X (gfx1250) — matrix operand order repaired and checked by reading, not compiled, not run on hardware
//
#include <hip/hip_runtime.h>

// ---------------------------------------------------------------------------
// 2-layer LSTM decoder for MI455X (gfx1250), persistent-kernel + WMMA f16.
//   B=64, T=256, H_DIM=512, H_UNITS=1024, gates=4096
// FP16 weights/activations (28MB, L2-resident), FP32 accumulators via
// v_wmma_f32_16x16x32_f16. 256 waves (32 WG x 8 waves); one wave owns a
// 16(batch) x 16(unit) tile and its 4 gate tiles; 2 grid barriers per step;
// double-buffered hidden state; cell state privately owned.
//  - Weight pointers laundered per step (asm volatile) so LICM cannot hoist
//    weight fragments across the t loop (avoids >256-VGPR spills).
//  - Fragment addressing shaped for GVS mode: uniform base advanced by k0
//    (SALU), fixed per-lane 32-bit offset (VGPR), per-gate displacement in
//    the instruction immediate -> no per-iteration 64-bit VALU adds.
//  - tanh via 2*sigmoid(2x)-1 (fast __expf path) in the gate nonlinearity.
// ---------------------------------------------------------------------------

#define TSTEPS 256
#define BATCH  64
#define HDIM   512
#define HUNITS 1024

typedef __attribute__((ext_vector_type(16))) _Float16 v16h;
typedef __attribute__((ext_vector_type(8)))  _Float16 v8h;
typedef __attribute__((ext_vector_type(8)))  float    v8f;

// ---- workspace layout (byte offsets, all 256B aligned) --------------------
#define OFF_XSEQ 0u                         // [T][B][512] f16   16,777,216 B
#define OFF_W0I  16777216u                  // [4096][512] f16    4,194,304 B
#define OFF_W0H  20971520u                  // [4096][1024] f16   8,388,608 B
#define OFF_W1I  29360128u                  // [4096][1024] f16   8,388,608 B
#define OFF_W1H  37748736u                  // [4096][1024] f16   8,388,608 B
#define OFF_B0   46137344u                  // [4096] f32            16,384 B
#define OFF_B1   46153728u                  // [4096] f32            16,384 B
#define OFF_H0   46170112u                  // 2x[64][1024] f16     262,144 B
#define OFF_H1   46432256u                  // 2x[64][1024] f16     262,144 B
#define OFF_C0   46694400u                  // [64][1024] f32       262,144 B
#define OFF_C1   46956544u                  // [64][1024] f32       262,144 B
#define OFF_BAR  47218688u                  // barrier counters

// Opaque zero: compiler must assume it changes every call, so anything
// offset by it is not loop-invariant (defeats LICM/spilling across t).
__device__ __forceinline__ int opaque_zero() {
  int z = 0;
  asm volatile("" : "+v"(z));
  return z;
}

__device__ __forceinline__ float sigf(float x) { return 1.0f / (1.0f + __expf(-x)); }
// tanh(x) = 2*sigmoid(2x) - 1 : one fast exp instead of libm tanhf
__device__ __forceinline__ float tanh_fast(float x) {
  return 2.0f / (1.0f + __expf(-2.0f * x)) - 1.0f;
}

__device__ __forceinline__ v16h make_a(v8h lo, v8h hi) {
  return __builtin_shufflevector(lo, hi, 0,1,2,3,4,5,6,7,8,9,10,11,12,13,14,15);
}

// ---------------------------------------------------------------------------
// One LSTM layer for one (mb, ub) tile owned by this wave.
//   gates = Ax[64xKx] @ Wi^T + Hp[64x1024] @ Wh^T + bias
//
// WMMA fragment layout (16-bit, 16x16x32) used here (contiguous k; A and B match):
//  A (16x32 MxK): lane L -> m=L&15; lanes 0-15 hold K=0..15,
//                 lanes 16-31 hold K=16..31, contiguous K per lane
//  B (32x16 KxN): lane L -> n=L&15; lanes 0-15 hold K=0..15,
//                 lanes 16-31 hold K=16..31, contiguous K per lane
// ---------------------------------------------------------------------------
__device__ __forceinline__ void lstm_layer_tile(
    const _Float16* __restrict__ Ax, int Kx,
    const _Float16* __restrict__ Hp,
    const _Float16* __restrict__ Wi,   // [4096][Kx]   (pre-laundered)
    const _Float16* __restrict__ Wh,   // [4096][1024] (pre-laundered)
    const float*    __restrict__ bias, // [4096] = bih + bhh
    float*          __restrict__ c,    // [64][1024]
    _Float16*       __restrict__ Hn,   // [64][1024] (double-buffer write)
    float*          __restrict__ outp, // nullptr or d_out + t*HUNITS
    int mb, int ub, int lane)
{
  v8f acc0 = {}, acc1 = {}, acc2 = {}, acc3 = {};
  const int r0 = mb * 16;
  const int j0 = ub * 16;

  // fixed per-lane element offsets (VGPR, loop-invariant)
  const int mn  = lane & 15;
  const int kbA = (lane >> 4) << 4;
  const int kbB = (lane >> 4) << 4;
  const int offA_x = (r0 + mn) * Kx     + kbA;   // into x_t
  const int offA_h = (r0 + mn) * HUNITS + kbA;   // into h_prev
  const int offB_i = (j0 + mn) * Kx     + kbB;   // into Wi (gate 0)
  const int offB_h = (j0 + mn) * HUNITS + kbB;   // into Wh (gate 0)

  // input-projection part: K = Kx ; gate g displaced by g*HUNITS*Kx elements
#pragma unroll 2
  for (int k0 = 0; k0 < Kx; k0 += 32) {
    const _Float16* Ak = Ax + k0;   // uniform (SALU advance)
    const _Float16* Wk = Wi + k0;   // uniform (SALU advance)
    v16h a  = make_a(*(const v8h*)(Ak + offA_x), *(const v8h*)(Ak + offA_x + 8));
    v16h b0 = *(const v16h*)(Wk + offB_i + 0 * HUNITS * Kx);
    v16h b1 = *(const v16h*)(Wk + offB_i + 1 * HUNITS * Kx);
    v16h b2 = *(const v16h*)(Wk + offB_i + 2 * HUNITS * Kx);
    v16h b3 = *(const v16h*)(Wk + offB_i + 3 * HUNITS * Kx);
    acc0 = __builtin_amdgcn_wmma_f32_16x16x32_f16(false, a, false, b0, (short)0, acc0, false, false);
    acc1 = __builtin_amdgcn_wmma_f32_16x16x32_f16(false, a, false, b1, (short)0, acc1, false, false);
    acc2 = __builtin_amdgcn_wmma_f32_16x16x32_f16(false, a, false, b2, (short)0, acc2, false, false);
    acc3 = __builtin_amdgcn_wmma_f32_16x16x32_f16(false, a, false, b3, (short)0, acc3, false, false);
  }
  // recurrent part: K = 1024 ; gate g displaced by g*HUNITS*HUNITS elements
#pragma unroll 2
  for (int k0 = 0; k0 < HUNITS; k0 += 32) {
    const _Float16* Ak = Hp + k0;
    const _Float16* Wk = Wh + k0;
    v16h a  = make_a(*(const v8h*)(Ak + offA_h), *(const v8h*)(Ak + offA_h + 8));
    v16h b0 = *(const v16h*)(Wk + offB_h + 0 * HUNITS * HUNITS);
    v16h b1 = *(const v16h*)(Wk + offB_h + 1 * HUNITS * HUNITS);
    v16h b2 = *(const v16h*)(Wk + offB_h + 2 * HUNITS * HUNITS);
    v16h b3 = *(const v16h*)(Wk + offB_h + 3 * HUNITS * HUNITS);
    acc0 = __builtin_amdgcn_wmma_f32_16x16x32_f16(false, a, false, b0, (short)0, acc0, false, false);
    acc1 = __builtin_amdgcn_wmma_f32_16x16x32_f16(false, a, false, b1, (short)0, acc1, false, false);
    acc2 = __builtin_amdgcn_wmma_f32_16x16x32_f16(false, a, false, b2, (short)0, acc2, false, false);
    acc3 = __builtin_amdgcn_wmma_f32_16x16x32_f16(false, a, false, b3, (short)0, acc3, false, false);
  }

  // elementwise LSTM cell on the owned 16x16 tile.
  // Acc layout: VGPR r, lane L -> row m = r + 8*(L>>4), col n = L&15.
  const int u = j0 + mn;
  const float bi = bias[u];
  const float bf = bias[u + 1 * HUNITS];
  const float bg = bias[u + 2 * HUNITS];
  const float bo = bias[u + 3 * HUNITS];
#pragma unroll
  for (int r = 0; r < 8; ++r) {
    const int b = r0 + r + ((lane >> 4) << 3);
    const float gi = acc0[r] + bi;
    const float gf = acc1[r] + bf;
    const float gg = acc2[r] + bg;
    const float go = acc3[r] + bo;
    const float cold = c[b * HUNITS + u];
    const float cn = sigf(gf) * cold + sigf(gi) * tanh_fast(gg);
    const float hn = sigf(go) * tanh_fast(cn);
    c[b * HUNITS + u]  = cn;
    Hn[b * HUNITS + u] = (_Float16)hn;
    if (outp) outp[(size_t)b * TSTEPS * HUNITS + u] = hn;  // outp already +t*H
  }
}

// ---------------------------------------------------------------------------
// Grid-wide sense-reversing barrier (32 resident workgroups).
// ---------------------------------------------------------------------------
__device__ __forceinline__ void grid_barrier(unsigned* bar, unsigned nwg) {
  __threadfence();           // every thread: device-visible stores
  __syncthreads();
  if (threadIdx.x == 0) {
    unsigned gen = __hip_atomic_load(&bar[1], __ATOMIC_ACQUIRE, __HIP_MEMORY_SCOPE_AGENT);
    unsigned prev = __hip_atomic_fetch_add(&bar[0], 1u, __ATOMIC_ACQ_REL, __HIP_MEMORY_SCOPE_AGENT);
    if (prev == nwg - 1u) {
      __hip_atomic_store(&bar[0], 0u, __ATOMIC_RELAXED, __HIP_MEMORY_SCOPE_AGENT);
      __hip_atomic_fetch_add(&bar[1], 1u, __ATOMIC_RELEASE, __HIP_MEMORY_SCOPE_AGENT);
    } else {
      while (__hip_atomic_load(&bar[1], __ATOMIC_ACQUIRE, __HIP_MEMORY_SCOPE_AGENT) == gen)
        __builtin_amdgcn_s_sleep(2);
    }
    __threadfence();
  }
  __syncthreads();
}

// ---------------------------------------------------------------------------
// Persistent kernel: all 256 timesteps, both layers, 2 barriers/step.
// ---------------------------------------------------------------------------
__global__ void __launch_bounds__(256, 1)
lstm_persistent(const _Float16* __restrict__ xseq,
                const _Float16* __restrict__ w0i, const _Float16* __restrict__ w0h,
                const _Float16* __restrict__ w1i, const _Float16* __restrict__ w1h,
                const float* __restrict__ b0, const float* __restrict__ b1,
                _Float16* __restrict__ h0buf, _Float16* __restrict__ h1buf,
                float* __restrict__ c0, float* __restrict__ c1,
                unsigned* __restrict__ bar, float* __restrict__ out)
{
  const int lane = threadIdx.x & 31;
  const int wid  = (blockIdx.x << 3) | (threadIdx.x >> 5);  // 0..255
  const int ub   = wid >> 2;   // unit tile 0..63
  const int mb   = wid & 3;    // batch tile 0..3
  const unsigned nwg = gridDim.x;

#pragma unroll 1
  for (int t = 0; t < TSTEPS; ++t) {
    const _Float16* xt  = xseq + (size_t)t * BATCH * HDIM;
    const _Float16* h0r = h0buf + (size_t)(t & 1) * BATCH * HUNITS;
    _Float16*       h0w = h0buf + (size_t)((t + 1) & 1) * BATCH * HUNITS;
    const _Float16* h1r = h1buf + (size_t)(t & 1) * BATCH * HUNITS;
    _Float16*       h1w = h1buf + (size_t)((t + 1) & 1) * BATCH * HUNITS;

    // layer 0: gates = x_t @ Wih0^T + h0_prev @ Whh0^T + bias0
    {
      const int z = opaque_zero();   // weights not loop-invariant -> no LICM
      lstm_layer_tile(xt, HDIM, h0r, w0i + z, w0h + z, b0, c0, h0w,
                      nullptr, mb, ub, lane);
    }
    grid_barrier(bar, nwg);   // h0(t) fully written before layer 1 reads it

    // layer 1: gates = h0(t) @ Wih1^T + h1_prev @ Whh1^T + bias1
    {
      const int z = opaque_zero();
      lstm_layer_tile(h0w, HUNITS, h1r, w1i + z, w1h + z, b1, c1, h1w,
                      out + (size_t)t * HUNITS, mb, ub, lane);
    }
    grid_barrier(bar, nwg);   // h1(t) fully written before next step reads it
  }
}

// ---------------------------------------------------------------------------
// Prep: fp32->fp16 weights, fuse biases, embedding gather, zero state+barrier.
// ---------------------------------------------------------------------------
__global__ void lstm_prep(const int* __restrict__ y, const float* __restrict__ embed,
                          const float* __restrict__ Wih0, const float* __restrict__ Whh0,
                          const float* __restrict__ bih0, const float* __restrict__ bhh0,
                          const float* __restrict__ Wih1, const float* __restrict__ Whh1,
                          const float* __restrict__ bih1, const float* __restrict__ bhh1,
                          _Float16* xseq, _Float16* w0i, _Float16* w0h,
                          _Float16* w1i, _Float16* w1h, float* b0, float* b1,
                          _Float16* h0buf, _Float16* h1buf, float* c0, float* c1,
                          unsigned* bar)
{
  const size_t tid = (size_t)blockIdx.x * blockDim.x + threadIdx.x;
  const size_t nth = (size_t)gridDim.x * blockDim.x;

  for (size_t i = tid; i < (size_t)4096 * 512; i += nth)  w0i[i] = (_Float16)Wih0[i];
  for (size_t i = tid; i < (size_t)4096 * 1024; i += nth) w0h[i] = (_Float16)Whh0[i];
  for (size_t i = tid; i < (size_t)4096 * 1024; i += nth) w1i[i] = (_Float16)Wih1[i];
  for (size_t i = tid; i < (size_t)4096 * 1024; i += nth) w1h[i] = (_Float16)Whh1[i];
  for (size_t i = tid; i < 4096; i += nth) { b0[i] = bih0[i] + bhh0[i]; b1[i] = bih1[i] + bhh1[i]; }

  // xseq[t][b][k] = embed[y[b][t]][k]
  for (size_t i = tid; i < (size_t)TSTEPS * BATCH * HDIM; i += nth) {
    const int t = (int)(i / (BATCH * HDIM));
    const int r = (int)(i % (BATCH * HDIM));
    const int b = r / HDIM;
    const int k = r % HDIM;
    const int row = y[b * TSTEPS + t];
    xseq[i] = (_Float16)embed[(size_t)row * HDIM + k];
  }

  for (size_t i = tid; i < (size_t)2 * BATCH * HUNITS; i += nth) {
    h0buf[i] = (_Float16)0.0f;
    h1buf[i] = (_Float16)0.0f;
  }
  for (size_t i = tid; i < (size_t)BATCH * HUNITS; i += nth) { c0[i] = 0.0f; c1[i] = 0.0f; }
  if (tid == 0) { bar[0] = 0u; bar[1] = 0u; }
}

// ---------------------------------------------------------------------------
extern "C" void kernel_launch(void* const* d_in, const int* in_sizes, int n_in,
                              void* d_out, int out_size, void* d_ws, size_t ws_size,
                              hipStream_t stream) {
  (void)in_sizes; (void)n_in; (void)out_size; (void)ws_size;
  const int*   y     = (const int*)  d_in[0];
  const float* embed = (const float*)d_in[1];
  const float* Wih0  = (const float*)d_in[2];
  const float* Whh0  = (const float*)d_in[3];
  const float* bih0  = (const float*)d_in[4];
  const float* bhh0  = (const float*)d_in[5];
  const float* Wih1  = (const float*)d_in[6];
  const float* Whh1  = (const float*)d_in[7];
  const float* bih1  = (const float*)d_in[8];
  const float* bhh1  = (const float*)d_in[9];

  char* ws = (char*)d_ws;
  _Float16* xseq  = (_Float16*)(ws + OFF_XSEQ);
  _Float16* w0i   = (_Float16*)(ws + OFF_W0I);
  _Float16* w0h   = (_Float16*)(ws + OFF_W0H);
  _Float16* w1i   = (_Float16*)(ws + OFF_W1I);
  _Float16* w1h   = (_Float16*)(ws + OFF_W1H);
  float*    b0    = (float*)   (ws + OFF_B0);
  float*    b1    = (float*)   (ws + OFF_B1);
  _Float16* h0buf = (_Float16*)(ws + OFF_H0);
  _Float16* h1buf = (_Float16*)(ws + OFF_H1);
  float*    c0    = (float*)   (ws + OFF_C0);
  float*    c1    = (float*)   (ws + OFF_C1);
  unsigned* bar   = (unsigned*)(ws + OFF_BAR);

  lstm_prep<<<2048, 256, 0, stream>>>(y, embed, Wih0, Whh0, bih0, bhh0,
                                      Wih1, Whh1, bih1, bhh1,
                                      xseq, w0i, w0h, w1i, w1h, b0, b1,
                                      h0buf, h1buf, c0, c1, bar);

  lstm_persistent<<<32, 256, 0, stream>>>(xseq, w0i, w0h, w1i, w1h, b0, b1,
                                          h0buf, h1buf, c0, c1, bar,
                                          (float*)d_out);
}
